// BinomialLoss_13941463843008
// MI455X (gfx1250) — matrix operand order repaired and checked by reading, not compiled, not run on hardware
//
#include <hip/hip_runtime.h>

#define N 8192
#define D 128

static constexpr float ALPHA  = 40.0f;
static constexpr float BETA   = 2.0f;
static constexpr float MARGIN = 0.5f;

typedef __attribute__((ext_vector_type(16))) __bf16 v16bf;
typedef __attribute__((ext_vector_type(8)))  __bf16 v8bf;
typedef __attribute__((ext_vector_type(8)))  float  v8f;

union BF16x16 { v16bf v; v8bf h[2]; };

// monotonic float -> uint key (for LDS atomic min/max over floats)
__device__ __forceinline__ unsigned fkey(float f) {
  unsigned u = __float_as_uint(f);
  return (u & 0x80000000u) ? ~u : (u | 0x80000000u);
}
__device__ __forceinline__ float funkey(unsigned k) {
  unsigned u = (k & 0x80000000u) ? (k ^ 0x80000000u) : ~k;
  return __uint_as_float(u);
}

// ---------------- fp32 -> bf16 (RNE) conversion into workspace ----------------
__global__ void convert_bf16_kernel(const float* __restrict__ x,
                                    unsigned short* __restrict__ xb, int n) {
  int i = blockIdx.x * blockDim.x + threadIdx.x;
  if (i < n) {
    unsigned u = __float_as_uint(x[i]);
    unsigned r = (u + 0x7FFFu + ((u >> 16) & 1u)) >> 16;
    xb[i] = (unsigned short)r;
  }
}

// ---------------- fused BinomialLoss kernel ----------------
// One workgroup (256 threads = 8 waves) owns 16 rows. Three sweeps over all
// column tiles, recomputing sim each time via v_wmma_f32_16x16x32_bf16
// (X in bf16 is 2 MB -> L2-resident; recompute beats materializing the
// 256 MB sim matrix by ~768 MB of HBM traffic):
//   sweep 1: hardest positive (min) / hardest negative (max) per row
//   sweep 2: keep-counts per row (needs sweep-1 row stats)
//   sweep 3: elementwise loss + grad, NT-streamed to HBM
__global__ __launch_bounds__(256) void binomial_loss_kernel(
    const unsigned short* __restrict__ xb_raw,
    const int* __restrict__ tgt,
    float* __restrict__ out_loss,
    float* __restrict__ out_grad) {
  const __bf16* __restrict__ Xb = reinterpret_cast<const __bf16*>(xb_raw);

  __shared__ unsigned sMin[16];
  __shared__ unsigned sMax[16];
  __shared__ int sP[16];
  __shared__ int sN[16];
  __shared__ int sTi[16];

  const int tid   = threadIdx.x;
  const int lane  = tid & 31;
  const int wid   = tid >> 5;        // 0..7 waves
  const int hf    = lane >> 4;       // lane half: 0 or 1
  const int lcol  = lane & 15;       // column / row-within-half selector
  const int ibase = blockIdx.x * 16;

  if (tid < 16) {
    sMin[tid] = 0xFFFFFFFFu;   // key(+inf)
    sMax[tid] = 0u;            // key(-inf)
    sP[tid] = 0;
    sN[tid] = 0;
    sTi[tid] = tgt[ibase + tid];
  }
  __syncthreads();

  // ---- A fragments: 16 x 128 bf16, resident in VGPRs for the whole kernel.
  // 16-bit A layout: lane L holds row (L&15); lanes<16 get K {c*32+0..7, c*32+16..23},
  // lanes>=16 get K {c*32+8..15, c*32+24..31}.
  BF16x16 A[4];
  {
    const __bf16* ap = Xb + (size_t)(ibase + lcol) * D + hf * 8;
#pragma unroll
    for (int c = 0; c < 4; ++c) {
      A[c].h[0] = *(const v8bf*)(ap + c * 32);
      A[c].h[1] = *(const v8bf*)(ap + c * 32 + 16);
    }
  }

  int tiR[8];
#pragma unroll
  for (int r = 0; r < 8; ++r) tiR[r] = sTi[r + hf * 8];

  // sim tile for columns [jbase, jbase+16): 4 chained WMMAs over K=128.
  // B layout (32x16 bf16): lane L holds column (L&15); lanes<16 K {c*32+0..7, c*32+16..23},
  // lanes>=16 K {c*32+8..15, c*32+24..31} (same as A).
  auto sim_tile = [&](int jbase) -> v8f {
    const __bf16* bp = Xb + (size_t)(jbase + lcol) * D + hf * 8;
    v8f c = {};
#pragma unroll
    for (int cc = 0; cc < 4; ++cc) {
      BF16x16 B;
      B.h[0] = *(const v8bf*)(bp + cc * 32);
      B.h[1] = *(const v8bf*)(bp + cc * 32 + 16);
      c = __builtin_amdgcn_wmma_f32_16x16x32_bf16(false, A[cc].v, false, B.v,
                                                  (short)0, c, false, false);
    }
    return c;
  };
  // C/D layout: c[r] on lane L is sim[ibase + r + 8*(L>>4)][jbase + (L&15)]

  const int iloc0 = ibase + hf * 8;   // global row of this lane's r=0 element

  // ---------------- sweep 1: hardest pos / hardest neg ----------------
  // compare-and-select update (v_cmp + s_and + v_cndmask): no fmin/fmax ->
  // no NaN-canonicalize ops; values are provably finite here.
  float mp[8], mn[8];
#pragma unroll
  for (int r = 0; r < 8; ++r) { mp[r] = 1e30f; mn[r] = -1e30f; }

  for (int jt = wid; jt < N / 16; jt += 8) {
    const int jbase = jt * 16;
    const v8f c = sim_tile(jbase);
    const int j  = jbase + lcol;
    const int tj = tgt[j];
#pragma unroll
    for (int r = 0; r < 8; ++r) {
      const float s   = c[r];
      const bool same = (tiR[r] == tj);
      const bool pos  = same & (j != (iloc0 + r));
      const bool updp = pos & (s < mp[r]);
      const bool updn = (!same) & (s > mn[r]);
      mp[r] = updp ? s : mp[r];
      mn[r] = updn ? s : mn[r];
    }
  }
#pragma unroll
  for (int r = 0; r < 8; ++r) {
    atomicMin(&sMin[r + hf * 8], fkey(mp[r]));
    atomicMax(&sMax[r + hf * 8], fkey(mn[r]));
  }
  __syncthreads();

  float minPos[8], maxNeg[8];
#pragma unroll
  for (int r = 0; r < 8; ++r) {
    minPos[r] = funkey(sMin[r + hf * 8]);
    maxNeg[r] = funkey(sMax[r + hf * 8]);
  }

  // ---------------- sweep 2: keep-counts (branchless) ----------------
  int pc[8], nc[8];
#pragma unroll
  for (int r = 0; r < 8; ++r) { pc[r] = 0; nc[r] = 0; }

  for (int jt = wid; jt < N / 16; jt += 8) {
    const int jbase = jt * 16;
    const v8f c = sim_tile(jbase);
    const int j  = jbase + lcol;
    const int tj = tgt[j];
#pragma unroll
    for (int r = 0; r < 8; ++r) {
      const float s   = c[r];
      const bool same = (tiR[r] == tj);
      const int posk = (int)(same & (j != (iloc0 + r)) & (s - 0.1f < maxNeg[r]));
      const int negk = (int)((!same) & (s + 0.1f > minPos[r]));
      pc[r] += posk;
      nc[r] += negk;
    }
  }
#pragma unroll
  for (int r = 0; r < 8; ++r) {
    atomicAdd(&sP[r + hf * 8], pc[r]);
    atomicAdd(&sN[r + hf * 8], nc[r]);
  }
  __syncthreads();

  float pgs[8], ngs[8];
  bool validR[8];
#pragma unroll
  for (int r = 0; r < 8; ++r) {
    const int P = sP[r + hf * 8];
    const int Q = sN[r + hf * 8];
    validR[r] = (P >= 1) && (Q >= 1);
    pgs[r] = -2.0f / (float)max(P, 1);
    ngs[r] =  2.0f / (float)max(Q, 1);
  }

  // ---------------- sweep 3: loss + grad, NT-streamed to HBM ----------------
  for (int jt = wid; jt < N / 16; jt += 8) {
    const int jbase = jt * 16;
    const v8f c = sim_tile(jbase);
    const int j  = jbase + lcol;
    const int tj = tgt[j];
#pragma unroll
    for (int r = 0; r < 8; ++r) {
      const float s   = c[r];
      const bool same = (tiR[r] == tj);
      const bool wp = same & (j != (iloc0 + r)) & (s - 0.1f < maxNeg[r]) & validR[r];
      const bool wn = (!same) & (s + 0.1f > minPos[r]) & validR[r];

      const float zp = -BETA * (s - MARGIN);
      const float zn = ALPHA * (s - MARGIN);
      // share exp between softplus and sigmoid; raw v_rcp_f32 for 1/(1+e)
      const float ep  = __expf(zp);
      const float en  = __expf(zn);
      const float rp  = __builtin_amdgcn_rcpf(1.0f + ep);
      const float rn  = __builtin_amdgcn_rcpf(1.0f + en);
      const float pos_loss = (2.0f / BETA)  * __logf(1.0f + ep);
      const float neg_loss = (2.0f / ALPHA) * __logf(1.0f + en);
      const float pos_grad = pgs[r] * (ep * rp);   // sigmoid(zp) = ep/(1+ep)
      const float neg_grad = ngs[r] * (en * rn);   // sigmoid(zn) = en/(1+en)

      const float loss = (wp ? pos_loss : 0.0f) + (wn ? neg_loss : 0.0f);
      const float grad = (wp ? pos_grad : 0.0f) + (wn ? neg_grad : 0.0f);

      const size_t idx = (size_t)(iloc0 + r) * (size_t)N + (size_t)j;
      __builtin_nontemporal_store(loss, out_loss + idx);
      __builtin_nontemporal_store(grad, out_grad + idx);
    }
  }
}

extern "C" void kernel_launch(void* const* d_in, const int* in_sizes, int n_in,
                              void* d_out, int out_size, void* d_ws, size_t ws_size,
                              hipStream_t stream) {
  const float* x   = (const float*)d_in[0];
  const int*   tgt = (const int*)d_in[1];
  float* out_loss = (float*)d_out;
  float* out_grad = out_loss + (size_t)N * (size_t)N;
  unsigned short* xb = (unsigned short*)d_ws;   // 8192*128 bf16 = 2 MB

  const int nelem = N * D;
  convert_bf16_kernel<<<(nelem + 255) / 256, 256, 0, stream>>>(x, xb, nelem);
  binomial_loss_kernel<<<N / 16, 256, 0, stream>>>(xb, tgt, out_loss, out_grad);
}
